// MultiHeadAttention_80805514707008
// MI455X (gfx1250) — matrix operand order repaired and checked by reading, not compiled, not run on hardware
//
#include <hip/hip_runtime.h>
#include <hip/hip_bf16.h>

#define D_MODEL 1024
#define NUM_HEADS 16
#define DEPTH 64
#define BATCH 2
#define SEQ 2048
#define TOK (BATCH * SEQ) /* 4096 */

typedef _Float16 h8v  __attribute__((ext_vector_type(8)));
typedef _Float16 h16v __attribute__((ext_vector_type(16)));
typedef float    f4v  __attribute__((ext_vector_type(4)));
typedef float    f8v  __attribute__((ext_vector_type(8)));

union H16U { h16v v; h8v h[2]; };

// ---- WMMA fragment loaders (CDNA5 wave32 layouts, CDNA5 ISA §7.12.2) ----

// A-matrix 16x32 f16: lane holds row (lane&15); halves 0..7 = K g*8..g*8+7,
// halves 8..15 = K 16+g*8..+7 (g = lane>>4). Two 16B loads, 16 elements apart.
__device__ __forceinline__ h16v frag_a(const _Float16* base, int ld, int lane) {
    const int r = lane & 15, g = (lane >> 4) & 1;
    const _Float16* p = base + (size_t)r * ld + g * 8;
    H16U u;
    u.h[0] = *(const h8v*)(p);
    u.h[1] = *(const h8v*)(p + 16);
    return u.v;
}

// B-matrix 32x16 f16: lane holds column (lane&15); elements 0..7 = K g*8..g*8+7, 8..15 = K 16+g*8..+7.
// Source is row-major [N][K] (i.e. B^T), so two 16B loads per lane, 16 elements apart.
__device__ __forceinline__ h16v frag_b(const _Float16* base, int ld, int lane) {
    const int c = lane & 15, g = (lane >> 4) & 1;
    const _Float16* p = base + (size_t)c * ld + g * 8;
    H16U u;
    u.h[0] = *(const h8v*)(p);
    u.h[1] = *(const h8v*)(p + 16);
    return u.v;
}

__device__ __forceinline__ f8v wmma16(h16v a, h16v b, f8v c) {
    return __builtin_amdgcn_wmma_f32_16x16x32_f16(false, a, false, b, (short)0, c,
                                                  false, false);
}

// ---------------- f32 -> f16 conversion (8 elems/thread, b128 in/out) ----------------
__global__ void __launch_bounds__(256) cvt_f16_kernel(const float* __restrict__ in,
                                                      _Float16* __restrict__ out, int n) {
    const int i = (blockIdx.x * 256 + threadIdx.x) * 8;
    if (i + 7 < n) {
        f4v x0 = *(const f4v*)(in + i);
        f4v x1 = *(const f4v*)(in + i + 4);
        h8v o;
#pragma unroll
        for (int e = 0; e < 4; ++e) {
            o[e]     = (_Float16)x0[e];
            o[e + 4] = (_Float16)x1[e];
        }
        *(h8v*)(out + i) = o;
    }
}

// ---------------- GEMM: C[M][1024] = A[M][1024] * W[1024][1024]^T + bias ----------------
// 32x64 tile per wave, software-pipelined (double-buffered fragments).
// MODE 0: f16 row-major out, val=(acc+bias)*scale
// MODE 1: f16 V head-transpose out: [B][H][64][S]
// MODE 2: f32 row-major out (final projection to d_out)
template <int MODE>
__global__ void __launch_bounds__(32) gemm_wmma_kernel(
    const _Float16* __restrict__ A, const _Float16* __restrict__ W,
    const float* __restrict__ bias, void* __restrict__ out, float scale) {
    constexpr int K = D_MODEL, N = D_MODEL;
    const int lane = threadIdx.x;
    const int c = lane & 15, g = (lane >> 4) & 1;
    const int m0 = blockIdx.x * 32;
    const int n0 = blockIdx.y * 64;
    const _Float16* pa0 = A + (size_t)m0 * K;
    const _Float16* pa1 = pa0 + (size_t)16 * K;
    const _Float16* pw0 = W + (size_t)n0 * K;

    // prologue: fragments for k0 = 0
    h16v a0 = frag_a(pa0, K, lane);
    h16v a1 = frag_a(pa1, K, lane);
    h16v bf[4];
#pragma unroll
    for (int j = 0; j < 4; ++j) bf[j] = frag_b(pw0 + (size_t)(j * 16) * K, K, lane);

    f8v acc[2][4] = {};
#pragma unroll 1
    for (int k0 = 32; k0 < K; k0 += 32) {
        // prefetch next K-step into fresh registers (overlaps with WMMAs below)
        h16v na0 = frag_a(pa0 + k0, K, lane);
        h16v na1 = frag_a(pa1 + k0, K, lane);
        h16v nbf[4];
#pragma unroll
        for (int j = 0; j < 4; ++j)
            nbf[j] = frag_b(pw0 + (size_t)(j * 16) * K + k0, K, lane);
#pragma unroll
        for (int j = 0; j < 4; ++j) {
            acc[0][j] = wmma16(a0, bf[j], acc[0][j]);
            acc[1][j] = wmma16(a1, bf[j], acc[1][j]);
        }
        a0 = na0;
        a1 = na1;
#pragma unroll
        for (int j = 0; j < 4; ++j) bf[j] = nbf[j];
    }
    // epilogue K-step
#pragma unroll
    for (int j = 0; j < 4; ++j) {
        acc[0][j] = wmma16(a0, bf[j], acc[0][j]);
        acc[1][j] = wmma16(a1, bf[j], acc[1][j]);
    }

#pragma unroll
    for (int j = 0; j < 4; ++j) {
        const int n = n0 + j * 16 + c;
        const float bv = bias[n];
#pragma unroll
        for (int mi = 0; mi < 2; ++mi) {
#pragma unroll
            for (int i = 0; i < 8; ++i) {
                const int row = m0 + mi * 16 + i + g * 8;  // C layout: VGPR i -> M=i (+8 hi half)
                const float val = (acc[mi][j][i] + bv) * scale;
                if (MODE == 0) {
                    ((_Float16*)out)[(size_t)row * N + n] = (_Float16)val;
                } else if (MODE == 1) {
                    const int b = row >> 11, s = row & 2047;
                    ((_Float16*)out)[(size_t)b * ((size_t)N * SEQ) + (size_t)n * SEQ + s] =
                        (_Float16)val;
                } else {
                    ((float*)out)[(size_t)row * N + n] = val;
                }
            }
        }
    }
}

// ---------------- flash attention: one wave per (b, h, 32-query tile) ----------------
// Qp: [TOK][1024] f16 pre-scaled by log2(e)/8. Kp: [TOK][1024] f16.
// Vt: [B][H][64][S] f16. mask: [B][1][S][S] f32. O: [TOK][1024] f16.
// 32 queries/wave: K & V fragments feed 16 WMMAs per 32-key step (2x reuse).
#define MASKC (-1.4426950408889634e9f) /* -1e9 * log2(e): exp2 domain */

__global__ void __launch_bounds__(32) attn_wmma_kernel(
    const _Float16* __restrict__ Qp, const _Float16* __restrict__ Kp,
    const _Float16* __restrict__ Vt, const float* __restrict__ mask,
    _Float16* __restrict__ O) {
    __shared__ alignas(16) _Float16 pt[2][2][16 * 32];  // [lds buf][q-subtile][16x32 P]

    const int lane = threadIdx.x;
    const int c = lane & 15, g = (lane >> 4) & 1;
    const int h = blockIdx.y, b = blockIdx.z;
    const int qb = blockIdx.x * 32;

    h16v aq[2][2];
#pragma unroll
    for (int u = 0; u < 2; ++u) {
        const _Float16* qbase = Qp + ((size_t)(b * SEQ + qb + u * 16)) * D_MODEL + h * DEPTH;
        aq[u][0] = frag_a(qbase, D_MODEL, lane);       // d 0..31
        aq[u][1] = frag_a(qbase + 32, D_MODEL, lane);  // d 32..63
    }

    f8v acc[2][4] = {};
    float m[2][8], l[2][8];
#pragma unroll
    for (int u = 0; u < 2; ++u)
#pragma unroll
        for (int i = 0; i < 8; ++i) { m[u][i] = -__builtin_inff(); l[u][i] = 0.0f; }

    const size_t maskbase = (size_t)b * SEQ * SEQ;
    const _Float16* vbase = Vt + ((size_t)(b * NUM_HEADS + h)) * DEPTH * SEQ;

#pragma unroll 1
    for (int ks0 = 0; ks0 < SEQ; ks0 += 32) {
        const int buf = (ks0 >> 5) & 1;
        // ---- K fragments for both 16-key subtiles (shared across both q-subtiles) ----
        h16v bk[2][2];
#pragma unroll
        for (int sub = 0; sub < 2; ++sub) {
            const _Float16* kb =
                Kp + ((size_t)(b * SEQ + ks0 + sub * 16)) * D_MODEL + h * DEPTH;
            bk[sub][0] = frag_b(kb, D_MODEL, lane);
            bk[sub][1] = frag_b(kb + 32, D_MODEL, lane);
        }
        // ---- logits (exp2 domain): 4 tiles of 16x16 ----
        f8v t[2][2];
#pragma unroll
        for (int u = 0; u < 2; ++u)
#pragma unroll
            for (int sub = 0; sub < 2; ++sub) {
                f8v z = {};
                f8v tt = wmma16(aq[u][0], bk[sub][0], z);
                tt = wmma16(aq[u][1], bk[sub][1], tt);
#pragma unroll
                for (int i = 0; i < 8; ++i) {
                    const int qrow = qb + u * 16 + i + g * 8;
                    tt[i] += mask[maskbase + (size_t)qrow * SEQ + ks0 + sub * 16 + c] * MASKC;
                }
                t[u][sub] = tt;
            }
        // ---- online softmax + P -> LDS (double-buffered, A-frag re-layout) ----
#pragma unroll
        for (int u = 0; u < 2; ++u) {
            float fs[8];
#pragma unroll
            for (int i = 0; i < 8; ++i) {
                float tm = fmaxf(t[u][0][i], t[u][1][i]);
                tm = fmaxf(tm, __shfl_xor(tm, 1));
                tm = fmaxf(tm, __shfl_xor(tm, 2));
                tm = fmaxf(tm, __shfl_xor(tm, 4));
                tm = fmaxf(tm, __shfl_xor(tm, 8));
                const float mn = fmaxf(m[u][i], tm);
                fs[i] = exp2f(m[u][i] - mn);
                m[u][i] = mn;
                const float p0 = exp2f(t[u][0][i] - mn);
                const float p1 = exp2f(t[u][1][i] - mn);
                l[u][i] = l[u][i] * fs[i] + p0 + p1;
                pt[buf][u][(i + g * 8) * 32 + c] = (_Float16)p0;
                pt[buf][u][(i + g * 8) * 32 + 16 + c] = (_Float16)p1;
            }
#pragma unroll
            for (int j = 0; j < 4; ++j)
#pragma unroll
                for (int i = 0; i < 8; ++i) acc[u][j][i] *= fs[i];
        }
        __syncthreads();  // single-wave workgroup: barrier ~free; orders LDS store->load
        const h16v ap0 = frag_a(&pt[buf][0][0], 32, lane);
        const h16v ap1 = frag_a(&pt[buf][1][0], 32, lane);

        // ---- acc += P (16x32) * V (32x64), V fragments shared across q-subtiles ----
#pragma unroll
        for (int j = 0; j < 4; ++j) {
            const h16v bv = frag_b(vbase + (size_t)(j * 16) * SEQ + ks0, SEQ, lane);
            acc[0][j] = wmma16(ap0, bv, acc[0][j]);
            acc[1][j] = wmma16(ap1, bv, acc[1][j]);
        }
    }

    // ---- finalize: divide by row sums, store merged-head f16 ----
#pragma unroll
    for (int u = 0; u < 2; ++u) {
#pragma unroll
        for (int i = 0; i < 8; ++i) {
            float s = l[u][i];
            s += __shfl_xor(s, 1);
            s += __shfl_xor(s, 2);
            s += __shfl_xor(s, 4);
            s += __shfl_xor(s, 8);
            l[u][i] = 1.0f / s;
        }
#pragma unroll
        for (int j = 0; j < 4; ++j)
#pragma unroll
            for (int i = 0; i < 8; ++i) {
                const int row = b * SEQ + qb + u * 16 + i + g * 8;
                const int col = h * DEPTH + j * 16 + c;
                O[(size_t)row * D_MODEL + col] = (_Float16)(acc[u][j][i] * l[u][i]);
            }
    }
}

// ---------------- host launcher ----------------
extern "C" void kernel_launch(void* const* d_in, const int* in_sizes, int n_in,
                              void* d_out, int out_size, void* d_ws, size_t ws_size,
                              hipStream_t stream) {
    (void)in_sizes; (void)n_in; (void)out_size; (void)ws_size;
    const float* q    = (const float*)d_in[0];
    const float* k    = (const float*)d_in[1];
    const float* v    = (const float*)d_in[2];
    const float* mask = (const float*)d_in[3];
    const float* wq   = (const float*)d_in[4];
    const float* bq   = (const float*)d_in[5];
    const float* wk   = (const float*)d_in[6];
    const float* bk   = (const float*)d_in[7];
    const float* wv   = (const float*)d_in[8];
    const float* bv   = (const float*)d_in[9];
    const float* wo   = (const float*)d_in[10];
    const float* bo   = (const float*)d_in[11];

    const size_t NT = (size_t)TOK * D_MODEL;      // 4M elems
    const size_t NW = (size_t)D_MODEL * D_MODEL;  // 1M elems
    _Float16* ws  = (_Float16*)d_ws;
    _Float16* qf  = ws;
    _Float16* kf  = qf + NT;
    _Float16* vf  = kf + NT;
    _Float16* wqf = vf + NT;
    _Float16* wkf = wqf + NW;
    _Float16* wvf = wkf + NW;
    _Float16* wof = wvf + NW;
    _Float16* Qp  = wof + NW;
    _Float16* Kp  = Qp + NT;
    _Float16* Vt  = Kp + NT;
    _Float16* Of  = Vt + NT;  // total 32M f16 = 64 MB

    // 1) downconvert activations + weights to f16 (pure bandwidth, b128 in/out)
    {
        const int nt = (int)NT, nw = (int)NW;
        cvt_f16_kernel<<<nt / (256 * 8), 256, 0, stream>>>(q, qf, nt);
        cvt_f16_kernel<<<nt / (256 * 8), 256, 0, stream>>>(k, kf, nt);
        cvt_f16_kernel<<<nt / (256 * 8), 256, 0, stream>>>(v, vf, nt);
        cvt_f16_kernel<<<nw / (256 * 8), 256, 0, stream>>>(wq, wqf, nw);
        cvt_f16_kernel<<<nw / (256 * 8), 256, 0, stream>>>(wk, wkf, nw);
        cvt_f16_kernel<<<nw / (256 * 8), 256, 0, stream>>>(wv, wvf, nw);
        cvt_f16_kernel<<<nw / (256 * 8), 256, 0, stream>>>(wo, wof, nw);
    }

    // 2) Q/K/V projections. Q pre-scaled by log2(e)/sqrt(64) for exp2-domain softmax.
    const dim3 ggrid(TOK / 32, D_MODEL / 64);
    const float qscale = 1.4426950408889634f / 8.0f;
    gemm_wmma_kernel<0><<<ggrid, 32, 0, stream>>>(qf, wqf, bq, Qp, qscale);
    gemm_wmma_kernel<0><<<ggrid, 32, 0, stream>>>(kf, wkf, bk, Kp, 1.0f);
    gemm_wmma_kernel<1><<<ggrid, 32, 0, stream>>>(vf, wvf, bv, Vt, 1.0f);

    // 3) flash attention: grid = (S/32 q-tiles, heads, batch), one wave per block
    attn_wmma_kernel<<<dim3(SEQ / 32, NUM_HEADS, BATCH), 32, 0, stream>>>(Qp, Kp, Vt, mask, Of);

    // 4) output projection -> f32 d_out
    gemm_wmma_kernel<2><<<ggrid, 32, 0, stream>>>(Of, wof, bo, (float*)d_out, 1.0f);
}
